// StackoverflowNet_53919019434250
// MI455X (gfx1250) — matrix operand order repaired and checked by reading, not compiled, not run on hardware
//
#include <hip/hip_runtime.h>

#define BB 256
#define TT 20
#define EE 96
#define HH 670
#define HP 672       // H padded to 32 (21 k-tiles)
#define VV 10004
#define H4 2680      // 4*H
#define H4P 2688     // padded to 16
#define KLSTM 768    // E+H=766 padded to 32 (24 k-tiles)
#define KP1 1440     // E+2H=1436 padded to 32 (45 k-tiles)

typedef __attribute__((ext_vector_type(16))) _Float16 v16h;
typedef __attribute__((ext_vector_type(8)))  float    v8f;

__device__ __forceinline__ float sigf(float x) { return 1.f / (1.f + __expf(-x)); }

// CDNA5 async global->LDS copy of one 16B chunk (tracked by ASYNCcnt)
__device__ __forceinline__ void async_cp16(unsigned lds_off, const void* gaddr) {
    asm volatile("global_load_async_to_lds_b128 %0, %1, off"
                 :: "v"(lds_off), "v"(gaddr) : "memory");
}
__device__ __forceinline__ void async_wait0() {
    asm volatile("s_wait_asynccnt 0x0" ::: "memory");
}

// ---------------- zero ----------------
__global__ void k_zero(float* p, int n) {
    int i = blockIdx.x * 256 + threadIdx.x;
    if (i < n) p[i] = 0.f;
}

// ------- pack weight [N,K] (optionally split K across two sources) into
// ------- CDNA5 WMMA f16 B-fragments: tile=(nt*Ktiles+kt), 32 lanes x 16 halfs.
__global__ void k_pack(const float* __restrict__ s0, const float* __restrict__ s1,
                       _Float16* __restrict__ dst, int Ntiles, int Ktiles,
                       int Nact, int K0, int K1) {
    int tid = blockIdx.x * 256 + threadIdx.x;
    int total = Ntiles * Ktiles * 32;
    if (tid >= total) return;
    int lane = tid & 31;
    int tile = tid >> 5;
    int kt = tile % Ktiles, nt = tile / Ktiles;
    int col = nt * 16 + (lane & 15);
    int kbase = kt * 32 + ((lane >> 4) << 3);
    int K = K0 + K1;
    _Float16* o = dst + (size_t)tile * 512 + lane * 16;
    for (int j = 0; j < 16; ++j) {
        int kk = kbase + (j & 7) + ((j & 8) << 1);
        float v = 0.f;
        if (col < Nact && kk < K)
            v = (kk < K0) ? s0[(size_t)col * K0 + kk] : s1[(size_t)col * K1 + (kk - K0)];
        o[j] = (_Float16)v;
    }
}

// ------- activation pack helper: (tile,lane) -> one 32B A-fragment -------
__device__ __forceinline__ void packA_one(const float* __restrict__ A, _Float16* __restrict__ dst,
                                          int tile, int lane, int Ktiles, int Kld) {
    int kt = tile % Ktiles, mT = tile / Ktiles;
    int row = lane & 15, hi = lane >> 4;
    const float* arow = A + (size_t)(mT * 16 + row) * Kld + kt * 32 + hi * 8;
    float4 a0 = *(const float4*)(arow);
    float4 a1 = *(const float4*)(arow + 4);
    float4 a2 = *(const float4*)(arow + 16);
    float4 a3 = *(const float4*)(arow + 20);
    v16h o;
    o[0] = (_Float16)a0.x;  o[1] = (_Float16)a0.y;  o[2] = (_Float16)a0.z;  o[3] = (_Float16)a0.w;
    o[4] = (_Float16)a1.x;  o[5] = (_Float16)a1.y;  o[6] = (_Float16)a1.z;  o[7] = (_Float16)a1.w;
    o[8] = (_Float16)a2.x;  o[9] = (_Float16)a2.y;  o[10] = (_Float16)a2.z; o[11] = (_Float16)a2.w;
    o[12] = (_Float16)a3.x; o[13] = (_Float16)a3.y; o[14] = (_Float16)a3.z; o[15] = (_Float16)a3.w;
    *(v16h*)(dst + (size_t)tile * 512 + lane * 16) = o;
}

__global__ void k_packA(const float* __restrict__ A, _Float16* __restrict__ dst,
                        int Mtiles, int Ktiles, int Kld) {
    int tid = blockIdx.x * 256 + threadIdx.x;
    if (tid >= Mtiles * Ktiles * 32) return;
    packA_one(A, dst, tid >> 5, tid & 31, Ktiles, Kld);
}

// pack two matrices in one launch
__global__ void k_packA2(const float* __restrict__ A0, const float* __restrict__ A1,
                         _Float16* __restrict__ d0, _Float16* __restrict__ d1,
                         int Mtiles, int Ktiles, int Kld) {
    int tid = blockIdx.x * 256 + threadIdx.x;
    int half = Mtiles * Ktiles * 32;
    if (tid >= 2 * half) return;
    int which = tid >= half;
    int t2 = which ? tid - half : tid;
    packA_one(which ? A1 : A0, which ? d1 : d0, t2 >> 5, t2 & 31, Ktiles, Kld);
}

// ------- register-blocked WMMA GEMM (p1) -------
template <int KT, int NB>
__global__ void k_gemm(const _Float16* __restrict__ Af, const _Float16* __restrict__ Bpk,
                       const float* __restrict__ bias, float* __restrict__ C,
                       int Mtiles, int nGroups, int Nact, int Nld, int relu) {
    int wid = (blockIdx.x * 256 + threadIdx.x) >> 5;
    if (wid >= Mtiles * nGroups) return;
    int mT = wid % Mtiles, ng = wid / Mtiles;
    int lane = threadIdx.x & 31;
    const _Float16* ap = Af + (size_t)mT * KT * 512 + lane * 16;
    const _Float16* bp = Bpk + (size_t)ng * NB * KT * 512 + lane * 16;
    v8f acc[NB] = {};
    for (int kt = 0; kt < KT; ++kt) {
        v16h a = *(const v16h*)ap; ap += 512;
#pragma unroll
        for (int u = 0; u < NB; ++u) {
            v16h b = *(const v16h*)(bp + (size_t)(u * KT + kt) * 512);
            acc[u] = __builtin_amdgcn_wmma_f32_16x16x32_f16(false, a, false, b, (short)0,
                                                            acc[u], false, false);
        }
    }
    int row = lane & 15, hi = lane >> 4;
#pragma unroll
    for (int u = 0; u < NB; ++u) {
        int n = (ng * NB + u) * 16 + row;
        if (n < Nact) {
            float bn = bias ? bias[n] : 0.f;
#pragma unroll
            for (int r = 0; r < 8; ++r) {
                float v = acc[u][r] + bn;
                if (relu) v = fmaxf(v, 0.f);
                C[(size_t)(mT * 16 + r + hi * 8) * Nld + n] = v;
            }
        }
    }
}

// ------- paired GEMM sharing A fragments (gl1 + ll1 in one launch) -------
template <int KT, int NB>
__global__ void k_gemm_pair(const _Float16* __restrict__ Af,
                            const _Float16* __restrict__ B0, const float* __restrict__ b0,
                            float* __restrict__ C0,
                            const _Float16* __restrict__ B1, const float* __restrict__ b1,
                            float* __restrict__ C1,
                            int Mtiles, int nGroups, int Nact, int Nld) {
    int wid = (blockIdx.x * 256 + threadIdx.x) >> 5;
    int half = Mtiles * nGroups;
    if (wid >= 2 * half) return;
    int which = wid >= half;
    int w2 = which ? wid - half : wid;
    const _Float16* Bpk = which ? B1 : B0;
    const float* bias = which ? b1 : b0;
    float* C = which ? C1 : C0;
    int mT = w2 % Mtiles, ng = w2 / Mtiles;
    int lane = threadIdx.x & 31;
    const _Float16* ap = Af + (size_t)mT * KT * 512 + lane * 16;
    const _Float16* bp = Bpk + (size_t)ng * NB * KT * 512 + lane * 16;
    v8f acc[NB] = {};
    for (int kt = 0; kt < KT; ++kt) {
        v16h a = *(const v16h*)ap; ap += 512;
#pragma unroll
        for (int u = 0; u < NB; ++u) {
            v16h b = *(const v16h*)(bp + (size_t)(u * KT + kt) * 512);
            acc[u] = __builtin_amdgcn_wmma_f32_16x16x32_f16(false, a, false, b, (short)0,
                                                            acc[u], false, false);
        }
    }
    int row = lane & 15, hi = lane >> 4;
#pragma unroll
    for (int u = 0; u < NB; ++u) {
        int n = (ng * NB + u) * 16 + row;
        if (n < Nact) {
            float bn = bias[n];
#pragma unroll
            for (int r = 0; r < 8; ++r)
                C[(size_t)(mT * 16 + r + hi * 8) * Nld + n] = acc[u][r] + bn;
        }
    }
}

// ------- dual LDS-staged gate GEMM: both LSTMs in one launch; B fragments
// ------- staged via CDNA5 async global->LDS copies, read back via ds_load_b128.
template <int KT, int NB>
__global__ void k_gemm_lds2(const _Float16* __restrict__ Af0, const _Float16* __restrict__ Af1,
                            const _Float16* __restrict__ B0, const _Float16* __restrict__ B1,
                            float* __restrict__ C0, float* __restrict__ C1,
                            int nGroups, int Nact, int Nld) {
    extern __shared__ _Float16 smem[];             // NB*KT*512 halfs
    int per = 2 * nGroups;                         // 2 mGroups x nGroups
    int which = blockIdx.x / per;
    int rem = blockIdx.x % per;
    int mGrp = rem / nGroups, nGrp = rem % nGroups;
    const _Float16* Af = which ? Af1 : Af0;
    const _Float16* Bpk = which ? B1 : B0;
    float* C = which ? C1 : C0;
    const _Float16* src = Bpk + (size_t)nGrp * NB * KT * 512;  // contiguous tile group
    const int n16 = NB * KT * 64;                  // 16B chunks
    for (int i = threadIdx.x; i < n16; i += 256)
        async_cp16((unsigned)(unsigned long long)(smem + (size_t)i * 8),
                   (const void*)(src + (size_t)i * 8));
    async_wait0();
    __syncthreads();
    int wave = threadIdx.x >> 5, lane = threadIdx.x & 31;
    int mT = mGrp * 8 + wave;
    const _Float16* ap = Af + (size_t)mT * KT * 512 + lane * 16;
    const _Float16* lp = smem + lane * 16;
    v8f acc[NB] = {};
    for (int kt = 0; kt < KT; ++kt) {
        v16h a = *(const v16h*)ap; ap += 512;
#pragma unroll
        for (int u = 0; u < NB; ++u) {
            v16h b = *(const v16h*)(lp + (size_t)(u * KT + kt) * 512);
            acc[u] = __builtin_amdgcn_wmma_f32_16x16x32_f16(false, a, false, b, (short)0,
                                                            acc[u], false, false);
        }
    }
    int row = lane & 15, hi = lane >> 4;
#pragma unroll
    for (int u = 0; u < NB; ++u) {
        int n = (nGrp * NB + u) * 16 + row;
        if (n < Nact) {
#pragma unroll
            for (int r = 0; r < 8; ++r)
                C[(size_t)(mT * 16 + r + hi * 8) * Nld + n] = acc[u][r];
        }
    }
}

// ------- fused final: logit = (xm*GL2+gb)*p2s0 + (xm*LL2+lb)*p2s1 -------
__global__ void k_final(const _Float16* __restrict__ Af,
                        const _Float16* __restrict__ Gpk, const _Float16* __restrict__ Lpk,
                        const float* __restrict__ gb, const float* __restrict__ lb,
                        const float* __restrict__ p2s, float* __restrict__ outbase,
                        int t, int staged) {
    const int KT = 3, NB = 2, nGroups = 313, Mtiles = 16;
    int wid = (blockIdx.x * 256 + threadIdx.x) >> 5;
    if (wid >= Mtiles * nGroups) return;
    int mT = wid % Mtiles, ng = wid / Mtiles;
    int lane = threadIdx.x & 31;
    const _Float16* ap = Af + (size_t)mT * KT * 512 + lane * 16;
    const _Float16* bg = Gpk + (size_t)ng * NB * KT * 512 + lane * 16;
    const _Float16* bl = Lpk + (size_t)ng * NB * KT * 512 + lane * 16;
    v8f ag[NB] = {}, al[NB] = {};
#pragma unroll
    for (int kt = 0; kt < KT; ++kt) {
        v16h a = *(const v16h*)ap; ap += 512;
#pragma unroll
        for (int u = 0; u < NB; ++u) {
            v16h g = *(const v16h*)(bg + (size_t)(u * KT + kt) * 512);
            v16h l = *(const v16h*)(bl + (size_t)(u * KT + kt) * 512);
            ag[u] = __builtin_amdgcn_wmma_f32_16x16x32_f16(false, a, false, g, (short)0,
                                                           ag[u], false, false);
            al[u] = __builtin_amdgcn_wmma_f32_16x16x32_f16(false, a, false, l, (short)0,
                                                           al[u], false, false);
        }
    }
    int row = lane & 15, hi = lane >> 4;
#pragma unroll
    for (int u = 0; u < NB; ++u) {
        int n = (ng * NB + u) * 16 + row;
        if (n < VV) {
            float gbn = gb[n], lbn = lb[n];
#pragma unroll
            for (int r = 0; r < 8; ++r) {
                int b = mT * 16 + r + hi * 8;
                float v = (ag[u][r] + gbn) * p2s[b * 2 + 0] + (al[u][r] + lbn) * p2s[b * 2 + 1];
                size_t o = staged ? ((size_t)b * VV + n)
                                  : ((size_t)b * VV * TT + (size_t)n * TT + t);
                outbase[o] = v;
            }
        }
    }
}

// ------- [T][B][V] staging -> [B,V,T] final layout, fully coalesced both ways -------
__global__ void k_tr(const float* __restrict__ tmp, float* __restrict__ out) {
    int idx = blockIdx.x * 256 + threadIdx.x;
    if (idx >= BB * VV) return;
    int b = idx / VV, v = idx % VV;
    float vals[TT];
#pragma unroll
    for (int t = 0; t < TT; ++t) vals[t] = tmp[(size_t)t * BB * VV + idx];
    float* o = out + (size_t)b * VV * TT + (size_t)v * TT;
#pragma unroll
    for (int i = 0; i < TT / 4; ++i)
        ((float4*)o)[i] = make_float4(vals[4 * i], vals[4 * i + 1], vals[4 * i + 2], vals[4 * i + 3]);
}

// ------- build u_g=[e|gh|0], u_l=[e|lh|0], stash e (states stride HP) -------
__global__ void k_build_u(const int* __restrict__ x, const float* __restrict__ emb,
                          const float* __restrict__ ghS, const float* __restrict__ lhS,
                          float* __restrict__ et, float* __restrict__ ug,
                          float* __restrict__ ul, int t) {
    int idx = blockIdx.x * 256 + threadIdx.x;
    if (idx >= BB * KLSTM) return;
    int b = idx / KLSTM, k = idx % KLSTM;
    float vg, vl;
    if (k < EE) {
        int xi = x[b * TT + t];
        float e = emb[(size_t)xi * EE + k];
        et[b * EE + k] = e;
        vg = e; vl = e;
    } else if (k < EE + HH) {
        vg = ghS[b * HP + (k - EE)];
        vl = lhS[b * HP + (k - EE)];
    } else { vg = 0.f; vl = 0.f; }
    ug[idx] = vg; ul[idx] = vl;
}

// ------- both LSTM pointwise updates in one launch (gate order i,f,g,o) -------
__global__ void k_lstm_both(const float* __restrict__ gg, const float* __restrict__ gl,
                            const float* __restrict__ g_bi, const float* __restrict__ g_bh,
                            const float* __restrict__ l_bi, const float* __restrict__ l_bh,
                            const float* __restrict__ gcS, float* __restrict__ gh_t,
                            float* __restrict__ gc_t, float* __restrict__ lhS,
                            float* __restrict__ lcS) {
    int idx = blockIdx.x * 256 + threadIdx.x;
    if (idx >= 2 * BB * HH) return;
    int which = idx >= BB * HH;
    int r = which ? idx - BB * HH : idx;
    int b = r / HH, h = r % HH;
    const float* g = (which ? gl : gg) + (size_t)b * H4P;
    const float* bi = which ? l_bi : g_bi;
    const float* bh = which ? l_bh : g_bh;
    float gi = g[h]          + bi[h]          + bh[h];
    float gf = g[HH + h]     + bi[HH + h]     + bh[HH + h];
    float gv = g[2 * HH + h] + bi[2 * HH + h] + bh[2 * HH + h];
    float go = g[3 * HH + h] + bi[3 * HH + h] + bh[3 * HH + h];
    float cin = which ? lcS[b * HP + h] : gcS[b * HP + h];
    float c = sigf(gf) * cin + sigf(gi) * tanhf(gv);
    float hv = sigf(go) * tanhf(c);
    if (which) { lcS[b * HP + h] = c; lhS[b * HP + h] = hv; }
    else       { gc_t[b * HH + h] = c; gh_t[b * HH + h] = hv; }
}

// ------- up = [e | gh_t | lh | 0] -------
__global__ void k_build_up(const float* __restrict__ et, const float* __restrict__ gh,
                           const float* __restrict__ lh, float* __restrict__ up) {
    int idx = blockIdx.x * 256 + threadIdx.x;
    if (idx >= BB * KP1) return;
    int b = idx / KP1, k = idx % KP1;
    float v = 0.f;
    if (k < EE) v = et[b * EE + k];
    else if (k < EE + HH) v = gh[b * HH + (k - EE)];
    else if (k < EE + 2 * HH) v = lh[b * HP + (k - EE - HH)];
    up[idx] = v;
}

// ------- fused p-MLP: p2..p4 -> pe softmax -> prob -> carry mix; one block per b -------
__global__ void k_pmlp(const float* __restrict__ h1,
                       const float* __restrict__ p2_w, const float* __restrict__ p2_b,
                       const float* __restrict__ p3_w, const float* __restrict__ p3_b,
                       const float* __restrict__ p4_w, const float* __restrict__ p4_b,
                       const float* __restrict__ pe_w, const float* __restrict__ pe_b,
                       float* __restrict__ prob_t,
                       const float* __restrict__ gh_t, const float* __restrict__ gc_t,
                       const float* __restrict__ lhS, const float* __restrict__ lcS,
                       float* __restrict__ ghS, float* __restrict__ gcS) {
    __shared__ float sh1[500], t2[250], t3[100], t4[50], pr[2];
    int b = blockIdx.x, tid = threadIdx.x;
    for (int i = tid; i < 500; i += 256) sh1[i] = h1[(size_t)b * 512 + i];
    __syncthreads();
    if (tid < 250) {
        float s = p2_b[tid];
        const float* w = p2_w + (size_t)tid * 500;
        for (int k = 0; k < 500; ++k) s += sh1[k] * w[k];
        t2[tid] = fmaxf(s, 0.f);
    }
    __syncthreads();
    if (tid < 100) {
        float s = p3_b[tid];
        const float* w = p3_w + (size_t)tid * 250;
        for (int k = 0; k < 250; ++k) s += t2[k] * w[k];
        t3[tid] = fmaxf(s, 0.f);
    }
    __syncthreads();
    if (tid < 50) {
        float s = p4_b[tid];
        const float* w = p4_w + (size_t)tid * 100;
        for (int k = 0; k < 100; ++k) s += t3[k] * w[k];
        t4[tid] = fmaxf(s, 0.f);
    }
    __syncthreads();
    if (tid == 0) {
        float l0 = pe_b[0], l1 = pe_b[1];
        for (int k = 0; k < 50; ++k) { l0 += t4[k] * pe_w[k]; l1 += t4[k] * pe_w[50 + k]; }
        float mx = fmaxf(l0, l1);
        float e0 = __expf(l0 - mx), e1 = __expf(l1 - mx);
        float s = e0 + e1;
        pr[0] = e0 / s; pr[1] = e1 / s;
        prob_t[b * 2 + 0] = pr[0]; prob_t[b * 2 + 1] = pr[1];
    }
    __syncthreads();
    float p0 = pr[0], p1 = pr[1];
    for (int h = tid; h < HH; h += 256) {
        ghS[b * HP + h] = p0 * lhS[b * HP + h] + p1 * gh_t[b * HH + h];
        gcS[b * HP + h] = p0 * lcS[b * HP + h] + p1 * gc_t[b * HH + h];
    }
}

// ------- fused q-chain: q1->q2->{q2e,q3}->{q3e,xm}; writes xm A-fragments directly -------
__global__ void k_qmlp(const float* __restrict__ sf,
                       const float* __restrict__ q1_w, const float* __restrict__ q1_b,
                       const float* __restrict__ q2_w, const float* __restrict__ q2_b,
                       const float* __restrict__ q2e_w, const float* __restrict__ q2e_b,
                       const float* __restrict__ q3_w, const float* __restrict__ q3_b,
                       const float* __restrict__ q3e_w, const float* __restrict__ q3e_b,
                       float* __restrict__ p1s_t, float* __restrict__ p2s_t,
                       _Float16* __restrict__ xmf) {
    __shared__ float ssf[192], t1[75], t2[30], t3[10], ps1[2];
    int b = blockIdx.x, tid = threadIdx.x;
    if (tid < 192) ssf[tid] = sf[(size_t)b * 192 + tid];
    __syncthreads();
    if (tid < 75) {
        float s = q1_b[tid];
        const float* w = q1_w + (size_t)tid * 192;
        for (int k = 0; k < 192; ++k) s += ssf[k] * w[k];
        t1[tid] = fmaxf(s, 0.f);
    }
    __syncthreads();
    if (tid < 30) {
        float s = q2_b[tid];
        const float* w = q2_w + (size_t)tid * 75;
        for (int k = 0; k < 75; ++k) s += t1[k] * w[k];
        t2[tid] = fmaxf(s, 0.f);
    }
    __syncthreads();
    if (tid == 0) {                       // q2e head -> p1s
        float l0 = q2e_b[0], l1 = q2e_b[1];
        for (int k = 0; k < 30; ++k) { l0 += t2[k] * q2e_w[k]; l1 += t2[k] * q2e_w[30 + k]; }
        l0 = fmaxf(l0, 0.f); l1 = fmaxf(l1, 0.f);
        float mx = fmaxf(l0, l1);
        float e0 = __expf(l0 - mx), e1 = __expf(l1 - mx);
        float s = e0 + e1;
        ps1[0] = e0 / s; ps1[1] = e1 / s;
        p1s_t[b * 2 + 0] = ps1[0]; p1s_t[b * 2 + 1] = ps1[1];
    }
    if (tid >= 32 && tid < 42) {          // q3 layer (10 wide)
        int n = tid - 32;
        float s = q3_b[n];
        const float* w = q3_w + (size_t)n * 30;
        for (int k = 0; k < 30; ++k) s += t2[k] * w[k];
        t3[n] = fmaxf(s, 0.f);
    }
    __syncthreads();
    if (tid == 0) {                       // q3e head -> p2s
        float l0 = q3e_b[0], l1 = q3e_b[1];
        for (int k = 0; k < 10; ++k) { l0 += t3[k] * q3e_w[k]; l1 += t3[k] * q3e_w[10 + k]; }
        l0 = fmaxf(l0, 0.f); l1 = fmaxf(l1, 0.f);
        float mx = fmaxf(l0, l1);
        float e0 = __expf(l0 - mx), e1 = __expf(l1 - mx);
        float s = e0 + e1;
        p2s_t[b * 2 + 0] = e0 / s; p2s_t[b * 2 + 1] = e1 / s;
    }
    if (tid < 96) {                       // xm = p1s0*gx + p1s1*lx -> A-fragment element
        float xv = ps1[0] * ssf[tid] + ps1[1] * ssf[96 + tid];
        int mT = b >> 4, row = b & 15, kt = tid >> 5, kk = tid & 31;
        int hi = (kk >> 3) & 1, j = (kk & 7) + ((kk & 16) ? 8 : 0);
        int lane = row + (hi << 4);
        xmf[((size_t)(mT * 3 + kt) * 32 + lane) * 16 + j] = (_Float16)xv;
    }
}

// ------- ph [B, T+2, T, 2] -------
__global__ void k_ph(const float* __restrict__ probs, const float* __restrict__ p1s,
                     const float* __restrict__ p2s, float* __restrict__ out) {
    int idx = blockIdx.x * 256 + threadIdx.x;
    int total = BB * (TT + 2) * TT * 2;
    if (idx >= total) return;
    int c = idx & 1;
    int j = (idx >> 1) % TT;
    int i = ((idx >> 1) / TT) % (TT + 2);
    int b = idx / (2 * TT * (TT + 2));
    float v;
    if (i < TT)       v = probs[(i * BB + b) * 2 + c];
    else if (i == TT) v = p1s[(j * BB + b) * 2 + c];
    else              v = p2s[(j * BB + b) * 2 + c];
    out[idx] = v;
}

extern "C" void kernel_launch(void* const* d_in, const int* in_sizes, int n_in,
                              void* d_out, int out_size, void* d_ws, size_t ws_size,
                              hipStream_t stream) {
    const int*   x     = (const int*)d_in[0];
    const float* emb   = (const float*)d_in[1];
    const float* g_wi  = (const float*)d_in[2];
    const float* g_wh  = (const float*)d_in[3];
    const float* g_bi  = (const float*)d_in[4];
    const float* g_bh  = (const float*)d_in[5];
    const float* l_wi  = (const float*)d_in[6];
    const float* l_wh  = (const float*)d_in[7];
    const float* l_bi  = (const float*)d_in[8];
    const float* l_bh  = (const float*)d_in[9];
    const float* gl1_w = (const float*)d_in[10];
    const float* gl1_b = (const float*)d_in[11];
    const float* ll1_w = (const float*)d_in[12];
    const float* ll1_b = (const float*)d_in[13];
    const float* gl2_w = (const float*)d_in[14];
    const float* gl2_b = (const float*)d_in[15];
    const float* ll2_w = (const float*)d_in[16];
    const float* ll2_b = (const float*)d_in[17];
    const float* p1_w  = (const float*)d_in[18];
    const float* p1_b  = (const float*)d_in[19];
    const float* p2_w  = (const float*)d_in[20];
    const float* p2_b  = (const float*)d_in[21];
    const float* p3_w  = (const float*)d_in[22];
    const float* p3_b  = (const float*)d_in[23];
    const float* p4_w  = (const float*)d_in[24];
    const float* p4_b  = (const float*)d_in[25];
    const float* pe_w  = (const float*)d_in[26];
    const float* pe_b  = (const float*)d_in[27];
    const float* q1_w  = (const float*)d_in[28];
    const float* q1_b  = (const float*)d_in[29];
    const float* q2_w  = (const float*)d_in[30];
    const float* q2_b  = (const float*)d_in[31];
    const float* q2e_w = (const float*)d_in[32];
    const float* q2e_b = (const float*)d_in[33];
    const float* q3_w  = (const float*)d_in[34];
    const float* q3_b  = (const float*)d_in[35];
    const float* q3e_w = (const float*)d_in[36];
    const float* q3e_b = (const float*)d_in[37];

    char* wsp = (char*)d_ws;
    auto alloc = [&](size_t bytes) -> char* {
        char* p = wsp;
        wsp += (bytes + 255) & ~(size_t)255;
        return p;
    };
    _Float16* Wg  = (_Float16*)alloc((size_t)168 * 24 * 512 * 2);
    _Float16* Wl  = (_Float16*)alloc((size_t)168 * 24 * 512 * 2);
    _Float16* P1  = (_Float16*)alloc((size_t)32 * 45 * 512 * 2);
    _Float16* GL1 = (_Float16*)alloc((size_t)6 * 21 * 512 * 2);
    _Float16* LL1 = (_Float16*)alloc((size_t)6 * 21 * 512 * 2);
    _Float16* GL2 = (_Float16*)alloc((size_t)626 * 3 * 512 * 2);
    _Float16* LL2 = (_Float16*)alloc((size_t)626 * 3 * 512 * 2);
    _Float16* ugf = (_Float16*)alloc((size_t)16 * 24 * 512 * 2);
    _Float16* ulf = (_Float16*)alloc((size_t)16 * 24 * 512 * 2);
    _Float16* upf = (_Float16*)alloc((size_t)16 * 45 * 512 * 2);
    _Float16* ghf = (_Float16*)alloc((size_t)16 * 21 * 512 * 2);
    _Float16* xmf = (_Float16*)alloc((size_t)16 * 3 * 512 * 2);
    float* states = (float*)alloc((size_t)4 * BB * HP * 4);
    float* ghS = states;
    float* gcS = states + BB * HP;
    float* lhS = states + 2 * BB * HP;
    float* lcS = states + 3 * BB * HP;
    float* gh_t    = (float*)alloc((size_t)BB * HH * 4);
    float* gc_t    = (float*)alloc((size_t)BB * HH * 4);
    float* et      = (float*)alloc((size_t)BB * EE * 4);
    float* ug      = (float*)alloc((size_t)BB * KLSTM * 4);
    float* ul      = (float*)alloc((size_t)BB * KLSTM * 4);
    float* gates_g = (float*)alloc((size_t)BB * H4P * 4);
    float* gates_l = (float*)alloc((size_t)BB * H4P * 4);
    float* up      = (float*)alloc((size_t)BB * KP1 * 4);
    float* h1      = (float*)alloc((size_t)BB * 512 * 4);
    float* probsA  = (float*)alloc((size_t)TT * BB * 2 * 4);
    float* p1sA    = (float*)alloc((size_t)TT * BB * 2 * 4);
    float* p2sA    = (float*)alloc((size_t)TT * BB * 2 * 4);
    float* sf      = (float*)alloc((size_t)BB * 192 * 4);

    size_t used = (size_t)(wsp - (char*)d_ws);
    size_t lbytes = (size_t)TT * BB * VV * 4;
    int staged = (ws_size >= used + lbytes + 1024) ? 1 : 0;
    float* ltmp = staged ? (float*)alloc(lbytes) : nullptr;

    auto blk = [](int n) { return (n + 255) / 256; };

    k_pack<<<blk(168 * 24 * 32), 256, 0, stream>>>(g_wi, g_wh, Wg, 168, 24, H4, EE, HH);
    k_pack<<<blk(168 * 24 * 32), 256, 0, stream>>>(l_wi, l_wh, Wl, 168, 24, H4, EE, HH);
    k_pack<<<blk(32 * 45 * 32), 256, 0, stream>>>(p1_w, nullptr, P1, 32, 45, 500, 1436, 0);
    k_pack<<<blk(6 * 21 * 32), 256, 0, stream>>>(gl1_w, nullptr, GL1, 6, 21, EE, HH, 0);
    k_pack<<<blk(6 * 21 * 32), 256, 0, stream>>>(ll1_w, nullptr, LL1, 6, 21, EE, HH, 0);
    k_pack<<<blk(626 * 3 * 32), 256, 0, stream>>>(gl2_w, nullptr, GL2, 626, 3, VV, EE, 0);
    k_pack<<<blk(626 * 3 * 32), 256, 0, stream>>>(ll2_w, nullptr, LL2, 626, 3, VV, EE, 0);
    k_zero<<<blk(4 * BB * HP), 256, 0, stream>>>(states, 4 * BB * HP);

    float* logits = (float*)d_out;
    float* ph_out = logits + (size_t)BB * VV * TT;
    const int LDSB = 4 * 24 * 512 * 2;   // 98304 B of B-fragments per block

    for (int t = 0; t < TT; ++t) {
        float* prob_t = probsA + (size_t)t * BB * 2;
        float* p1s_t  = p1sA + (size_t)t * BB * 2;
        float* p2s_t  = p2sA + (size_t)t * BB * 2;

        k_build_u<<<blk(BB * KLSTM), 256, 0, stream>>>(x, emb, ghS, lhS, et, ug, ul, t);
        k_packA2<<<blk(2 * 16 * 24 * 32), 256, 0, stream>>>(ug, ul, ugf, ulf, 16, 24, KLSTM);
        k_gemm_lds2<24, 4><<<2 * 2 * 42, 256, LDSB, stream>>>(ugf, ulf, Wg, Wl,
                                                              gates_g, gates_l, 42, H4, H4P);
        k_lstm_both<<<blk(2 * BB * HH), 256, 0, stream>>>(gates_g, gates_l, g_bi, g_bh,
                                                          l_bi, l_bh, gcS, gh_t, gc_t,
                                                          lhS, lcS);
        k_build_up<<<blk(BB * KP1), 256, 0, stream>>>(et, gh_t, lhS, up);
        k_packA<<<blk(16 * 45 * 32), 256, 0, stream>>>(up, upf, 16, 45, KP1);
        k_gemm<45, 4><<<blk(16 * 8 * 32), 256, 0, stream>>>(upf, P1, p1_b, h1,
                                                            16, 8, 500, 512, 1);
        k_pmlp<<<BB, 256, 0, stream>>>(h1, p2_w, p2_b, p3_w, p3_b, p4_w, p4_b, pe_w, pe_b,
                                       prob_t, gh_t, gc_t, lhS, lcS, ghS, gcS);
        k_packA<<<blk(16 * 21 * 32), 256, 0, stream>>>(ghS, ghf, 16, 21, HP);
        k_gemm_pair<21, 2><<<blk(2 * 16 * 3 * 32), 256, 0, stream>>>(ghf, GL1, gl1_b, sf,
                                                                     LL1, ll1_b, sf + 96,
                                                                     16, 3, EE, 192);
        k_qmlp<<<BB, 256, 0, stream>>>(sf, q1_w, q1_b, q2_w, q2_b, q2e_w, q2e_b,
                                       q3_w, q3_b, q3e_w, q3e_b, p1s_t, p2s_t, xmf);
        float* obase = staged ? (ltmp + (size_t)t * BB * VV) : logits;
        k_final<<<blk(16 * 313 * 32), 256, 0, stream>>>(xmf, GL2, LL2, gl2_b, ll2_b,
                                                        p2s_t, obase, t, staged);
    }
    if (staged) k_tr<<<blk(BB * VV), 256, 0, stream>>>(ltmp, logits);
    k_ph<<<blk(BB * (TT + 2) * TT * 2), 256, 0, stream>>>(probsA, p1sA, p2sA, ph_out);
}
